// RSAGE2_4105988735704
// MI455X (gfx1250) — matrix operand order repaired and checked by reading, not compiled, not run on hardware
//
#include <hip/hip_runtime.h>
#include <hip/hip_bf16.h>

#define NN 100000
#define EE 300000
#define DD 256
#define CC 40
#define MT (NN / 16)   // 6250 row tiles (N divisible by 16)

typedef __attribute__((ext_vector_type(16))) __bf16 v16bf;
typedef __attribute__((ext_vector_type(8)))  __bf16 v8bf;
typedef __attribute__((ext_vector_type(8)))  float  v8f;

static __device__ __forceinline__ v8f wmma_bf16(v16bf a, v16bf b, v8f c) {
  // D = A(16x32 bf16) * B(32x16 bf16) + C(16x16 f32)
  return __builtin_amdgcn_wmma_f32_16x16x32_bf16(false, a, false, b, (short)0, c,
                                                 false, false);
}

static __device__ __forceinline__ v8f vzero8() {
  v8f z = {0.f, 0.f, 0.f, 0.f, 0.f, 0.f, 0.f, 0.f};
  return z;
}

// Per-lane A fragment: lane<16 -> row=lane, K = {kt*32+0..7, kt*32+16..23}
//                      lane>=16 -> row=lane-16, K = {kt*32+8..15, kt*32+24..31}
static __device__ __forceinline__ v16bf load_a_frag(const __bf16* rowbase, int kb) {
  v8bf lo = *(const v8bf*)(rowbase + kb);
  v8bf hi = *(const v8bf*)(rowbase + kb + 16);
  v16bf r;
#pragma unroll
  for (int i = 0; i < 8; ++i) { r[i] = lo[i]; r[i + 8] = hi[i]; }
  return r;
}

// ---------------------------------------------------------------- utilities

__global__ void zero4_kernel(float4* p, long n4) {
  long i = (long)blockIdx.x * blockDim.x + threadIdx.x;
  if (i < n4) p[i] = float4{0.f, 0.f, 0.f, 0.f};
}

__global__ void cvt4_kernel(const float* __restrict__ in, __bf16* __restrict__ out, long n4) {
  long i = (long)blockIdx.x * blockDim.x + threadIdx.x;
  if (i >= n4) return;
  float4 v = ((const float4*)in)[i];
  long b = i * 4;
  out[b + 0] = (__bf16)v.x; out[b + 1] = (__bf16)v.y;
  out[b + 2] = (__bf16)v.z; out[b + 3] = (__bf16)v.w;
}

__global__ void count_kernel(const int* __restrict__ dst, float* __restrict__ cnt, int E) {
  int e = blockIdx.x * blockDim.x + threadIdx.x;
  if (e < E) atomicAdd(&cnt[dst[e]], 1.0f);
}

// 64 threads per edge, 4 floats each: msg = w * h[src] scatter-added into ssum[dst]
__global__ void scatter_kernel(const float* __restrict__ h, const int* __restrict__ src,
                               const int* __restrict__ dst, const float* __restrict__ w,
                               float* __restrict__ ssum, int E) {
  long t = (long)blockIdx.x * blockDim.x + threadIdx.x;
  int e = (int)(t >> 6);
  if (e >= E) return;
  int q = (int)(t & 63);
  int s = src[e], d = dst[e];
  float wt = w[e];
  float4 v = ((const float4*)(h + (size_t)s * DD))[q];
  float* o = ssum + (size_t)d * DD + q * 4;
  atomicAdd(o + 0, v.x * wt);
  atomicAdd(o + 1, v.y * wt);
  atomicAdd(o + 2, v.z * wt);
  atomicAdd(o + 3, v.w * wt);
}

// neigh = ssum / max(cnt,1), emitted directly as bf16 GEMM operand
__global__ void finalize_kernel(const float* __restrict__ ssum, const float* __restrict__ cnt,
                                __bf16* __restrict__ outb, long n4) {
  long i = (long)blockIdx.x * blockDim.x + threadIdx.x;
  if (i >= n4) return;
  int node = (int)(i >> 6);  // 64 float4 per 256-wide row
  float c = cnt[node];
  float r = 1.0f / (c > 1.0f ? c : 1.0f);
  float4 v = ((const float4*)ssum)[i];
  long b = i * 4;
  outb[b + 0] = (__bf16)(v.x * r); outb[b + 1] = (__bf16)(v.y * r);
  outb[b + 2] = (__bf16)(v.z * r); outb[b + 3] = (__bf16)(v.w * r);
}

// Pack K x Ncols f32 weight (optionally sum of two) into per-lane WMMA B-fragment
// layout: out[((kt*ntiles+nt)*32+lane)*16 + i], lane<16 -> K=kt*32+{0..7,16..23}[i], lane>=16 -> K=kt*32+{8..15,24..31}[i],
// n = nt*16 + (lane&15). Columns >= Ncols padded with zero.
__global__ void pack_b2_kernel(const float* __restrict__ W1, const float* __restrict__ W2,
                               __bf16* __restrict__ out, int Ncols, int ntiles, int total) {
  int idx = blockIdx.x * blockDim.x + threadIdx.x;
  if (idx >= total) return;
  int i = idx & 15;
  int lane = (idx >> 4) & 31;
  int frag = idx >> 9;
  int nt = frag % ntiles;
  int kt = frag / ntiles;
  int k = kt * 32 + (i & 8) * 2 + ((lane < 16) ? 0 : 8) + (i & 7);
  int n = nt * 16 + (lane & 15);
  float v = 0.f;
  if (n < Ncols) {
    v = W1[(size_t)k * Ncols + n];
    if (W2) v += W2[(size_t)k * Ncols + n];
  }
  out[idx] = (__bf16)v;
}

__global__ void brelsum_kernel(const float* __restrict__ brel, float* __restrict__ o) {
  int i = blockIdx.x * blockDim.x + threadIdx.x;
  if (i >= 2 * DD) return;
  int l = i >> 8, d = i & (DD - 1);
  o[i] = brel[l * 2 * DD + d] + brel[l * 2 * DD + DD + d];
}

// ------------------------------------------------- fused RGCN-SAGE layer GEMM
// h_new = h@Wskip + bskip + relu(h@WselfSum + n0@Wn0 + n1@Wn1 + brelsum)
__global__ __launch_bounds__(256) void rsage_layer_kernel(
    const __bf16* __restrict__ hb, const __bf16* __restrict__ n0b,
    const __bf16* __restrict__ n1b, const __bf16* __restrict__ WsP,
    const __bf16* __restrict__ Wn0P, const __bf16* __restrict__ Wn1P,
    const __bf16* __restrict__ WskP, const float* __restrict__ brelsum,
    const float* __restrict__ bskip, float* __restrict__ out_f,
    __bf16* __restrict__ out_b, int write_f) {
  __shared__ __bf16 ls[3][16][264];  // 264 pitch: 528B rows -> conflict-staggered banks
  const int mt = blockIdx.x;
  const size_t rowbase = (size_t)mt * 16 * DD;
  const __bf16* gsrc[3] = {hb + rowbase, n0b + rowbase, n1b + rowbase};
#pragma unroll
  for (int m = 0; m < 3; ++m) {
#pragma unroll
    for (int it = 0; it < 2; ++it) {
      int idx = threadIdx.x + it * 256;  // 512 x 16B chunks per 16x256 bf16 tile
      int r = idx >> 5, c = idx & 31;
      uint4 v = ((const uint4*)gsrc[m])[idx];
      *(uint4*)&ls[m][r][c * 8] = v;
    }
  }
  __syncthreads();

  const int wave = threadIdx.x >> 5;
  const int lane = threadIdx.x & 31;
  const int arow = lane & 15;
  const int aoff = (lane < 16) ? 0 : 8;

  v8f acc2[2] = {vzero8(), vzero8()};
  v8f accS[2] = {vzero8(), vzero8()};

#pragma unroll
  for (int kt = 0; kt < 8; ++kt) {
    const int kb = kt * 32 + aoff;
    v16bf aH = load_a_frag(&ls[0][arow][0], kb);
    v16bf a0 = load_a_frag(&ls[1][arow][0], kb);
    v16bf a1 = load_a_frag(&ls[2][arow][0], kb);
    if (kt < 7) {  // warm next K-step's weight fragments (global_prefetch_b8)
      size_t pf = ((size_t)((kt + 1) * 16 + wave) * 32 + lane) * 16;
      __builtin_prefetch(WsP + pf, 0, 1);
      __builtin_prefetch(WskP + pf, 0, 1);
    }
#pragma unroll
    for (int jj = 0; jj < 2; ++jj) {
      const int j = wave + jj * 8;
      const size_t f = ((size_t)(kt * 16 + j) * 32 + lane) * 16;
      acc2[jj] = wmma_bf16(aH, *(const v16bf*)(WsP + f), acc2[jj]);
      acc2[jj] = wmma_bf16(a0, *(const v16bf*)(Wn0P + f), acc2[jj]);
      acc2[jj] = wmma_bf16(a1, *(const v16bf*)(Wn1P + f), acc2[jj]);
      accS[jj] = wmma_bf16(aH, *(const v16bf*)(WskP + f), accS[jj]);
    }
  }

  const int rbase = mt * 16 + ((lane < 16) ? 0 : 8);
#pragma unroll
  for (int jj = 0; jj < 2; ++jj) {
    const int col = (wave + jj * 8) * 16 + (lane & 15);
    const float bs = bskip[col];
    const float br = brelsum[col];
#pragma unroll
    for (int i = 0; i < 8; ++i) {
      float h2 = acc2[jj][i] + br;
      float v = (accS[jj][i] + bs) + (h2 > 0.f ? h2 : 0.f);
      size_t o = (size_t)(rbase + i) * DD + col;
      if (write_f) out_f[o] = v;
      out_b[o] = (__bf16)v;
    }
  }
}

// ---------------------------------------------------------- fused MLP head
// logits = relu(h@Wc1 + bc1) @ Wc2 + bc2   (Wc2 packed 40->48 cols, zero padded)
__global__ __launch_bounds__(256) void head_kernel(
    const __bf16* __restrict__ hb, const __bf16* __restrict__ Wc1P,
    const float* __restrict__ bc1, const __bf16* __restrict__ Wc2P,
    const float* __restrict__ bc2, float* __restrict__ out) {
  __shared__ __bf16 lh[16][264];
  __shared__ __bf16 lt[16][264];
  const int mt = blockIdx.x;
  const size_t rowbase = (size_t)mt * 16 * DD;
#pragma unroll
  for (int it = 0; it < 2; ++it) {
    int idx = threadIdx.x + it * 256;
    int r = idx >> 5, c = idx & 31;
    uint4 v = ((const uint4*)(hb + rowbase))[idx];
    *(uint4*)&lh[r][c * 8] = v;
  }
  __syncthreads();

  const int wave = threadIdx.x >> 5;
  const int lane = threadIdx.x & 31;
  const int arow = lane & 15;
  const int aoff = (lane < 16) ? 0 : 8;
  const int rofs = (lane < 16) ? 0 : 8;

  v8f acc[2] = {vzero8(), vzero8()};
#pragma unroll
  for (int kt = 0; kt < 8; ++kt) {
    v16bf a = load_a_frag(&lh[arow][0], kt * 32 + aoff);
#pragma unroll
    for (int jj = 0; jj < 2; ++jj) {
      const int j = wave + jj * 8;
      const size_t f = ((size_t)(kt * 16 + j) * 32 + lane) * 16;
      acc[jj] = wmma_bf16(a, *(const v16bf*)(Wc1P + f), acc[jj]);
    }
  }
#pragma unroll
  for (int jj = 0; jj < 2; ++jj) {
    const int col = (wave + jj * 8) * 16 + (lane & 15);
    const float b1 = bc1[col];
#pragma unroll
    for (int i = 0; i < 8; ++i) {
      float v = acc[jj][i] + b1;
      lt[rofs + i][col] = (__bf16)(v > 0.f ? v : 0.f);
    }
  }
  __syncthreads();

  if (wave < 3) {  // wave-uniform: EXEC all ones inside (WMMA requirement)
    v8f accO = vzero8();
#pragma unroll
    for (int kt = 0; kt < 8; ++kt) {
      v16bf a = load_a_frag(&lt[arow][0], kt * 32 + aoff);
      const size_t f = ((size_t)(kt * 3 + wave) * 32 + lane) * 16;
      accO = wmma_bf16(a, *(const v16bf*)(Wc2P + f), accO);
    }
    const int col = wave * 16 + (lane & 15);
    if (col < CC) {
      const float b2 = bc2[col];
      const int rbase = mt * 16 + rofs;
#pragma unroll
      for (int i = 0; i < 8; ++i)
        out[(size_t)(rbase + i) * CC + col] = accO[i] + b2;
    }
  }
}

// -------------------------------------------------------------------- launch

extern "C" void kernel_launch(void* const* d_in, const int* in_sizes, int n_in,
                              void* d_out, int out_size, void* d_ws, size_t ws_size,
                              hipStream_t stream) {
  (void)in_sizes; (void)n_in; (void)out_size; (void)ws_size;
  const float* x    = (const float*)d_in[0];
  const int* src0   = (const int*)d_in[1];
  const int* dst0   = (const int*)d_in[2];
  const float* ew0  = (const float*)d_in[3];
  const int* src1   = (const int*)d_in[4];
  const int* dst1   = (const int*)d_in[5];
  const float* ew1  = (const float*)d_in[6];
  const float* Wself  = (const float*)d_in[7];   // [2,2,256,256]
  const float* Wneigh = (const float*)d_in[8];   // [2,2,256,256]
  const float* brel   = (const float*)d_in[9];   // [2,2,256]
  const float* Wskip  = (const float*)d_in[10];  // [2,256,256]
  const float* bskip  = (const float*)d_in[11];  // [2,256]
  const float* Wc1    = (const float*)d_in[12];  // [256,256]
  const float* bc1    = (const float*)d_in[13];  // [256]
  const float* Wc2    = (const float*)d_in[14];  // [256,40]
  const float* bc2    = (const float*)d_in[15];  // [40]
  float* out = (float*)d_out;

  char* wp = (char*)d_ws;
  auto alloc = [&](size_t bytes) -> char* {
    char* p = wp;
    wp += (bytes + 255) & ~(size_t)255;
    return p;
  };
  float* cnt0 = (float*)alloc((size_t)NN * 4);
  float* cnt1 = (float*)alloc((size_t)NN * 4);
  float* ssum = (float*)alloc((size_t)NN * DD * 4);
  float* hf   = (float*)alloc((size_t)NN * DD * 4);
  __bf16* hbA = (__bf16*)alloc((size_t)NN * DD * 2);
  __bf16* hbB = (__bf16*)alloc((size_t)NN * DD * 2);
  __bf16* n0b = (__bf16*)alloc((size_t)NN * DD * 2);
  __bf16* n1b = (__bf16*)alloc((size_t)NN * DD * 2);
  __bf16 *WsP[2], *Wn0P[2], *Wn1P[2], *WskP[2];
  for (int l = 0; l < 2; ++l) {
    WsP[l]  = (__bf16*)alloc((size_t)DD * DD * 2);
    Wn0P[l] = (__bf16*)alloc((size_t)DD * DD * 2);
    Wn1P[l] = (__bf16*)alloc((size_t)DD * DD * 2);
    WskP[l] = (__bf16*)alloc((size_t)DD * DD * 2);
  }
  __bf16* Wc1P = (__bf16*)alloc((size_t)DD * DD * 2);
  __bf16* Wc2P = (__bf16*)alloc((size_t)DD * 48 * 2);
  float* brs = (float*)alloc(2 * DD * 4);

  const int TB = 256;
  const long n4 = (long)NN * DD / 4;
  const int n4B = (int)((n4 + TB - 1) / TB);
  const int scB = (int)(((long)EE * 64 + TB - 1) / TB);

  // ---- weight prep (Wself summed over relations: h@W distributes) ----
  for (int l = 0; l < 2; ++l) {
    const float* Ws = Wself + (size_t)l * 2 * DD * DD;
    const float* Wn = Wneigh + (size_t)l * 2 * DD * DD;
    pack_b2_kernel<<<256, TB, 0, stream>>>(Ws, Ws + DD * DD, WsP[l], DD, 16, 65536);
    pack_b2_kernel<<<256, TB, 0, stream>>>(Wn, nullptr, Wn0P[l], DD, 16, 65536);
    pack_b2_kernel<<<256, TB, 0, stream>>>(Wn + DD * DD, nullptr, Wn1P[l], DD, 16, 65536);
    pack_b2_kernel<<<256, TB, 0, stream>>>(Wskip + (size_t)l * DD * DD, nullptr, WskP[l], DD, 16, 65536);
  }
  pack_b2_kernel<<<256, TB, 0, stream>>>(Wc1, nullptr, Wc1P, DD, 16, 65536);
  pack_b2_kernel<<<48, TB, 0, stream>>>(Wc2, nullptr, Wc2P, CC, 3, 12288);
  brelsum_kernel<<<2, TB, 0, stream>>>(brel, brs);

  // ---- degree counts (graph static across layers) ----
  zero4_kernel<<<(NN / 4 + TB - 1) / TB, TB, 0, stream>>>((float4*)cnt0, NN / 4);
  zero4_kernel<<<(NN / 4 + TB - 1) / TB, TB, 0, stream>>>((float4*)cnt1, NN / 4);
  count_kernel<<<(EE + TB - 1) / TB, TB, 0, stream>>>(dst0, cnt0, EE);
  count_kernel<<<(EE + TB - 1) / TB, TB, 0, stream>>>(dst1, cnt1, EE);

  // ---- x -> bf16 operand ----
  cvt4_kernel<<<n4B, TB, 0, stream>>>(x, hbA, n4);

  // ---- layer 0 ----
  zero4_kernel<<<n4B, TB, 0, stream>>>((float4*)ssum, n4);
  scatter_kernel<<<scB, TB, 0, stream>>>(x, src0, dst0, ew0, ssum, EE);
  finalize_kernel<<<n4B, TB, 0, stream>>>(ssum, cnt0, n0b, n4);
  zero4_kernel<<<n4B, TB, 0, stream>>>((float4*)ssum, n4);
  scatter_kernel<<<scB, TB, 0, stream>>>(x, src1, dst1, ew1, ssum, EE);
  finalize_kernel<<<n4B, TB, 0, stream>>>(ssum, cnt1, n1b, n4);
  rsage_layer_kernel<<<MT, TB, 0, stream>>>(hbA, n0b, n1b, WsP[0], Wn0P[0], Wn1P[0],
                                            WskP[0], brs, bskip, hf, hbB, 1);

  // ---- layer 1 ----
  zero4_kernel<<<n4B, TB, 0, stream>>>((float4*)ssum, n4);
  scatter_kernel<<<scB, TB, 0, stream>>>(hf, src0, dst0, ew0, ssum, EE);
  finalize_kernel<<<n4B, TB, 0, stream>>>(ssum, cnt0, n0b, n4);
  zero4_kernel<<<n4B, TB, 0, stream>>>((float4*)ssum, n4);
  scatter_kernel<<<scB, TB, 0, stream>>>(hf, src1, dst1, ew1, ssum, EE);
  finalize_kernel<<<n4B, TB, 0, stream>>>(ssum, cnt1, n1b, n4);
  rsage_layer_kernel<<<MT, TB, 0, stream>>>(hbB, n0b, n1b, WsP[1], Wn0P[1], Wn1P[1],
                                            WskP[1], brs + DD, bskip + DD, hf, hbA, 0);

  // ---- classifier head ----
  head_kernel<<<MT, TB, 0, stream>>>(hbA, Wc1P, bc1, Wc2P, bc2, out);
}
